// MyGRUTransition_64604898066782
// MI455X (gfx1250) — matrix operand order repaired and checked by reading, not compiled, not run on hardware
//
#include <hip/hip_runtime.h>
#include <hip/hip_bf16.h>

// Problem dims
#define BSZ   32768
#define SDIM  512
#define KDIM  512

typedef __attribute__((ext_vector_type(16))) __bf16 v16bf;
typedef __attribute__((ext_vector_type(8)))  __bf16 v8bf;
typedef __attribute__((ext_vector_type(8)))  float  v8f;

// ---- workspace layout (bytes) ----
#define OFF_WIG   0ull                                   // 1024*512 bf16 = 1 MB
#define OFF_WSG   (OFF_WIG   + 1024ull*512*2)            // 1 MB
#define OFF_WIS   (OFF_WSG   + 1024ull*512*2)            // 512 KB
#define OFF_WH2H  (OFF_WIS   +  512ull*512*2)            // 512 KB
#define OFF_DATA  (4ull<<20)                             // data bf16: 32 MB
#define OFF_STATE (OFF_DATA  + (size_t)BSZ*SDIM*2)       // state bf16: 32 MB
#define OFF_UPD   (OFF_STATE + (size_t)BSZ*SDIM*2)       // update fp32: 64 MB
#define OFF_IS    (OFF_UPD   + (size_t)BSZ*SDIM*4)       // input_state fp32: 64 MB
#define OFF_RST   (OFF_IS    + (size_t)BSZ*SDIM*4)       // rstate bf16: 32 MB

__device__ __forceinline__ float sigmoidf_(float x) {
  return 1.0f / (1.0f + __expf(-x));
}

__device__ __forceinline__ v8f wmma_bf16(v16bf a, v16bf b, v8f c) {
  // D(f32 16x16) = A(16x32 bf16) * B(32x16 bf16) + C
  return __builtin_amdgcn_wmma_f32_16x16x32_bf16(
      /*neg_a=*/false, a, /*neg_b=*/false, b,
      /*c_mod=*/(short)0, c, /*reuse_a=*/false, /*reuse_b=*/false);
}

// A fragment (16x32 bf16) from bf16 row-major A[*, lda].
// lane<16: row=rowBase+lane,    K = kb+{0..15}
// lane>=16: row=rowBase+lane-16, K = kb+{16..31}
__device__ __forceinline__ v16bf load_a(const __bf16* __restrict__ A, int lda,
                                        int rowBase, int kb, int lane) {
  const int r = rowBase + (lane & 15);
  const int k = kb + ((lane >> 4) << 4);
  const __bf16* p = A + (size_t)r * lda + k;
  v8bf lo = *(const v8bf*)(p);       // K k..k+7      (16B)
  v8bf hi = *(const v8bf*)(p + 8);   // K k+8..k+15   (16B)
  v16bf a;
#pragma unroll
  for (int i = 0; i < 8; ++i) { a[i] = lo[i]; a[8 + i] = hi[i]; }
  return a;
}

// B fragment (32x16 bf16) for D = X * W^T: B[k,n] = W[n,k], W row-major [N,K] bf16.
// lane<16: col=colBase+lane, K=kb..kb+15 (32B contiguous); lane>=16: K=kb+16..kb+31.
__device__ __forceinline__ v16bf load_b(const __bf16* __restrict__ W, int ldw,
                                        int colBase, int kb, int lane) {
  const int n = colBase + (lane & 15);
  const int k = kb + ((lane >> 4) << 4);
  return *(const v16bf*)(W + (size_t)n * ldw + k);
}

// Accumulate a 64x64 wave tile over K: 4x4 WMMA tiles, 16 wmma per K=32 step.
__device__ __forceinline__ void gemm64x64(v8f acc[4][4], const __bf16* __restrict__ A,
                                          const __bf16* __restrict__ W, int K,
                                          int rowBase, int colBase, int lane) {
#pragma unroll 2
  for (int kb = 0; kb < K; kb += 32) {
    v16bf a[4], b[4];
#pragma unroll
    for (int i = 0; i < 4; ++i) a[i] = load_a(A, K, rowBase + 16 * i, kb, lane);
#pragma unroll
    for (int j = 0; j < 4; ++j) b[j] = load_b(W, K, colBase + 16 * j, kb, lane);
#pragma unroll
    for (int i = 0; i < 4; ++i)
#pragma unroll
      for (int j = 0; j < 4; ++j)
        acc[i][j] = wmma_bf16(a[i], b[j], acc[i][j]);
  }
}

// fp32 -> bf16 conversion (vectorized by 4)
__global__ void cvt_f32_to_bf16(const float* __restrict__ src,
                                unsigned short* __restrict__ dst, int nquads) {
  int i = blockIdx.x * blockDim.x + threadIdx.x;
  if (i < nquads) {
    float4 v = ((const float4*)src)[i];
    ushort4 o;
    o.x = __builtin_bit_cast(unsigned short, (__bf16)v.x);
    o.y = __builtin_bit_cast(unsigned short, (__bf16)v.y);
    o.z = __builtin_bit_cast(unsigned short, (__bf16)v.z);
    o.w = __builtin_bit_cast(unsigned short, (__bf16)v.w);
    ((ushort4*)dst)[i] = o;
  }
}

// gate = data @ W_ig.T + b_ig + state @ W_sg.T   (N = 1024)
// cols < 512 : update[m,n] = sigmoid(gate)             (fp32)
// cols >= 512: rstate[m,c] = state[m,c]*sigmoid(gate)  (bf16), c = n-512
__global__ __launch_bounds__(256) void gru_gate_kernel(
    const unsigned short* __restrict__ data_u, const unsigned short* __restrict__ state_u,
    const float* __restrict__ state_f32,
    const unsigned short* __restrict__ wig_u, const unsigned short* __restrict__ wsg_u,
    const float* __restrict__ b_ig, float* __restrict__ upd,
    unsigned short* __restrict__ rst_u) {
  const __bf16* dataB  = (const __bf16*)data_u;
  const __bf16* stateB = (const __bf16*)state_u;
  const __bf16* Wig = (const __bf16*)wig_u;
  const __bf16* Wsg = (const __bf16*)wsg_u;
  __bf16* rst = (__bf16*)rst_u;
  const int lane = threadIdx.x & 31, wid = threadIdx.x >> 5;
  const int rowBase = blockIdx.x * 128 + (wid & 1) * 64;
  const int colBase = blockIdx.y * 256 + (wid >> 1) * 64;
  v8f acc[4][4] = {};
  gemm64x64(acc, dataB,  Wig, KDIM, rowBase, colBase, lane);  // data @ W_ig.T
  gemm64x64(acc, stateB, Wsg, KDIM, rowBase, colBase, lane);  // + state @ W_sg.T
#pragma unroll
  for (int ti = 0; ti < 4; ++ti)
#pragma unroll
    for (int tj = 0; tj < 4; ++tj)
#pragma unroll
      for (int v = 0; v < 8; ++v) {
        int m = rowBase + ti * 16 + ((lane >> 4) << 3) + v;
        int n = colBase + tj * 16 + (lane & 15);
        float g = acc[ti][tj][v] + b_ig[n];
        if (n < SDIM) {
          upd[(size_t)m * SDIM + n] = sigmoidf_(g);
        } else {
          int c = n - SDIM;
          float s0 = state_f32[(size_t)m * SDIM + c];
          rst[(size_t)m * SDIM + c] = (__bf16)(s0 * sigmoidf_(g));
        }
      }
}

// input_state = data @ W_is.T + b_is  (fp32)
__global__ __launch_bounds__(256) void gru_is_kernel(
    const unsigned short* __restrict__ data_u, const unsigned short* __restrict__ wis_u,
    const float* __restrict__ b_is, float* __restrict__ istate) {
  const __bf16* dataB = (const __bf16*)data_u;
  const __bf16* Wis = (const __bf16*)wis_u;
  const int lane = threadIdx.x & 31, wid = threadIdx.x >> 5;
  const int rowBase = blockIdx.x * 128 + (wid & 1) * 64;
  const int colBase = blockIdx.y * 256 + (wid >> 1) * 64;
  v8f acc[4][4] = {};
  gemm64x64(acc, dataB, Wis, KDIM, rowBase, colBase, lane);
#pragma unroll
  for (int ti = 0; ti < 4; ++ti)
#pragma unroll
    for (int tj = 0; tj < 4; ++tj)
#pragma unroll
      for (int v = 0; v < 8; ++v) {
        int m = rowBase + ti * 16 + ((lane >> 4) << 3) + v;
        int n = colBase + tj * 16 + (lane & 15);
        istate[(size_t)m * SDIM + n] = acc[ti][tj][v] + b_is[n];
      }
}

// h = rstate @ W_h2h.T ; out = u*tanh(input_state + h) + (1-u)*state
__global__ __launch_bounds__(256) void gru_h_kernel(
    const unsigned short* __restrict__ rst_u, const unsigned short* __restrict__ wh_u,
    const float* __restrict__ istate, const float* __restrict__ upd,
    const float* __restrict__ state, float* __restrict__ out) {
  const __bf16* rst = (const __bf16*)rst_u;
  const __bf16* Wh = (const __bf16*)wh_u;
  const int lane = threadIdx.x & 31, wid = threadIdx.x >> 5;
  const int rowBase = blockIdx.x * 128 + (wid & 1) * 64;
  const int colBase = blockIdx.y * 256 + (wid >> 1) * 64;
  v8f acc[4][4] = {};
  gemm64x64(acc, rst, Wh, SDIM, rowBase, colBase, lane);
#pragma unroll
  for (int ti = 0; ti < 4; ++ti)
#pragma unroll
    for (int tj = 0; tj < 4; ++tj)
#pragma unroll
      for (int v = 0; v < 8; ++v) {
        int m = rowBase + ti * 16 + ((lane >> 4) << 3) + v;
        int n = colBase + tj * 16 + (lane & 15);
        size_t idx = (size_t)m * SDIM + n;
        float sh = tanhf(istate[idx] + acc[ti][tj][v]);
        float u = upd[idx];
        out[idx] = u * sh + (1.0f - u) * state[idx];
      }
}

extern "C" void kernel_launch(void* const* d_in, const int* in_sizes, int n_in,
                              void* d_out, int out_size, void* d_ws, size_t ws_size,
                              hipStream_t stream) {
  const float* data  = (const float*)d_in[0];   // [B, 512]
  const float* state = (const float*)d_in[1];   // [B, 512]
  const float* W_is  = (const float*)d_in[2];   // [512, 512]
  const float* b_is  = (const float*)d_in[3];   // [512]
  const float* W_ig  = (const float*)d_in[4];   // [1024, 512]
  const float* b_ig  = (const float*)d_in[5];   // [1024]
  const float* W_sg  = (const float*)d_in[6];   // [1024, 512]
  const float* W_h2h = (const float*)d_in[7];   // [512, 512]
  float* out = (float*)d_out;

  char* ws = (char*)d_ws;
  unsigned short* wig  = (unsigned short*)(ws + OFF_WIG);
  unsigned short* wsg  = (unsigned short*)(ws + OFF_WSG);
  unsigned short* wis  = (unsigned short*)(ws + OFF_WIS);
  unsigned short* wh2h = (unsigned short*)(ws + OFF_WH2H);
  unsigned short* dataB  = (unsigned short*)(ws + OFF_DATA);
  unsigned short* stateB = (unsigned short*)(ws + OFF_STATE);
  float*          upd  = (float*)(ws + OFF_UPD);
  float*          ist  = (float*)(ws + OFF_IS);
  unsigned short* rst  = (unsigned short*)(ws + OFF_RST);

  // 1) fp32 -> bf16 prepass: weights (tiny, L2-resident) + activations
  int q;
  q = (1024 * 512) / 4;
  cvt_f32_to_bf16<<<(q + 255) / 256, 256, 0, stream>>>(W_ig, wig, q);
  cvt_f32_to_bf16<<<(q + 255) / 256, 256, 0, stream>>>(W_sg, wsg, q);
  q = (512 * 512) / 4;
  cvt_f32_to_bf16<<<(q + 255) / 256, 256, 0, stream>>>(W_is, wis, q);
  cvt_f32_to_bf16<<<(q + 255) / 256, 256, 0, stream>>>(W_h2h, wh2h, q);
  q = (BSZ * SDIM) / 4;
  cvt_f32_to_bf16<<<(q + 255) / 256, 256, 0, stream>>>(data, dataB, q);
  cvt_f32_to_bf16<<<(q + 255) / 256, 256, 0, stream>>>(state, stateB, q);

  // 2) gates (K accumulated over data then state), sigmoid epilogue
  gru_gate_kernel<<<dim3(BSZ / 128, 1024 / 256), 256, 0, stream>>>(
      dataB, stateB, state, wig, wsg, b_ig, upd, rst);

  // 3) input_state GEMM
  gru_is_kernel<<<dim3(BSZ / 128, SDIM / 256), 256, 0, stream>>>(
      dataB, wis, b_is, ist);

  // 4) h2h GEMM + tanh + blend -> new_state
  gru_h_kernel<<<dim3(BSZ / 128, SDIM / 256), 256, 0, stream>>>(
      rst, wh2h, ist, upd, state, out);
}
